// NNPDHG4DynMRIwTVNN_20856361189836
// MI455X (gfx1250) — matrix operand order repaired and checked by reading, not compiled, not run on hardware
//
#include <hip/hip_runtime.h>
#include <math.h>

// ---------------------------------------------------------------------------
// Problem geometry (from reference): Nb=Nz=1, Nc=10, Nt=12, U=Fd=128
// ---------------------------------------------------------------------------
#define NU 128
#define NFD 128
#define NT 12
#define NC 10
#define UF (NU*NFD)          // 16384
#define NX (NT*UF)           // 196608 complex elements (x-sized)
#define NY (NC*NX)           // 1966080 complex elements (y-sized)
#define NSLICE (NC*NT)       // 120 FFT slices per A/AH apply
#define TPDHG 64
#define CGIT 6

typedef __attribute__((ext_vector_type(16))) _Float16 v16h;
typedef __attribute__((ext_vector_type(8)))  float    v8f;
typedef __attribute__((ext_vector_type(4)))  unsigned uv4;
typedef __attribute__((ext_vector_type(8)))  int      iv8;
typedef __attribute__((ext_vector_type(4)))  int      iv4;

// Tensor Data Mover availability: builtin exists on both toolchains but with
// different arity (ROCm7.2/clang-22: 5 args, amdgpu-toolchain/clang-23: 6).
#if defined(__HIP_DEVICE_COMPILE__) && defined(__has_builtin)
# if __has_builtin(__builtin_amdgcn_tensor_load_to_lds)
#  if __clang_major__ >= 23
#   define TDM_ARITY 6
#  else
#   define TDM_ARITY 5
#  endif
# else
#  define TDM_ARITY 0
# endif
#else
# define TDM_ARITY 0
#endif

static __device__ __forceinline__ float2 f2(float a, float b){ float2 r; r.x=a; r.y=b; return r; }
static __device__ __forceinline__ float2 f2add(float2 a,float2 b){ return f2(a.x+b.x,a.y+b.y); }
static __device__ __forceinline__ float2 f2sub(float2 a,float2 b){ return f2(a.x-b.x,a.y-b.y); }
static __device__ __forceinline__ float2 f2scale(float2 a,float s){ return f2(a.x*s,a.y*s); }
static __device__ __forceinline__ float2 cmul(float2 a,float2 b){ return f2(a.x*b.x-a.y*b.y, a.x*b.y+a.y*b.x); }
static __device__ __forceinline__ float2 cmulConjA(float2 a,float2 b){ return f2(a.x*b.x+a.y*b.y, a.x*b.y-a.y*b.x); }
static __device__ __forceinline__ float sigmoidf_(float x){ return 1.f/(1.f+expf(-x)); }
static __device__ __forceinline__ float softplusf_(float x){ return (x>20.f)?x:log1pf(expf(x)); }

static __device__ __forceinline__ unsigned packh(float a, float b){
  union { _Float16 h[2]; unsigned u; } p;
  p.h[0] = (_Float16)a; p.h[1] = (_Float16)b; return p.u;
}

#if TDM_ARITY > 0
// 2-D TDM tile load, descriptor per CDNA5 ISA 8.3/8.4. All dims in DWORD
// (data_size=4B) units. Pad feature reproduces the bank-conflict-free LDS
// strides (padIv: 3->16dw interval, padAmt: code N -> N+1 dwords).
static __device__ __forceinline__ void tdmLoad2D(unsigned ldsAddr, const void* gptr,
    unsigned td0, unsigned td1, unsigned tile0, unsigned tile1,
    unsigned st0, unsigned padIv, unsigned padAmt)
{
  unsigned long long ga = (unsigned long long)(size_t)gptr;
  unsigned st1 = td0 * td1;      // dim1 stride (unused for 2-D tiles)
  uv4 g0;
  g0[0] = 1u;                                        // count=1, user descriptor
  g0[1] = ldsAddr;                                   // lds_addr (bytes)
  g0[2] = (unsigned)(ga & 0xFFFFFFFFull);            // global_addr[31:0]
  g0[3] = (unsigned)((ga >> 32) & 0x01FFFFFFull) | (2u << 30);  // addr[56:32] | type=2
  iv8 g1;
  g1[0] = (int)((2u << 16) | (1u << 20) | (padIv << 22) | (padAmt << 25)); // data_size=4B, pad_enable
  g1[1] = (int)((td0 & 0xFFFFu) << 16);              // tensor_dim0[15:0]
  g1[2] = (int)((td0 >> 16) | ((td1 & 0xFFFFu) << 16));
  g1[3] = (int)((td1 >> 16) | (tile0 << 16));        // tile_dim0
  g1[4] = (int)tile1;                                // tile_dim1 (tile_dim2=0)
  g1[5] = (int)st0;                                  // tensor_dim0_stride[31:0]
  g1[6] = (int)((st1 & 0xFFFFu) << 16);              // dim0_stride[47:32]=0 | dim1_stride[15:0]
  g1[7] = (int)(st1 >> 16);                          // dim1_stride[47:16]
  iv4 z4 = {};
#if TDM_ARITY == 6
  iv8 z8 = {};
  __builtin_amdgcn_tensor_load_to_lds(g0, g1, z4, z4, z8, 0);
#else
  __builtin_amdgcn_tensor_load_to_lds(g0, g1, z4, z4, 0);
#endif
}
#endif

// ---------------------------------------------------------------------------
// Core: batched 128x128x128 complex GEMM  C = scale*(A*B)  via WMMA f16->f32.
// 512 threads = 16 wave32s; wave w owns row strip (w>>1)*16 x 4 column tiles.
// LDS panels hold K=32 slabs, planar re/im, pre-packed into the 32-bit K-pairs
// the WMMA fragments need (one ds_load_b32 per fragment VGPR).
// The DFT-matrix operand (aIsW/bIsW) is staged from pre-packed f16 planes in
// global memory -- via the Tensor Data Mover when available (pad feature
// recreates the padded LDS strides), else a packed u32 copy.
// ---------------------------------------------------------------------------
__global__ __launch_bounds__(512)
void cgemm128(const float2* __restrict__ Ab, long aStride,
              const float2* __restrict__ Bb, long bStride,
              float2* __restrict__ Cb, long cStride, float scale,
              const unsigned* __restrict__ pkARe, const unsigned* __restrict__ pkAIm, int aIsW,
              const unsigned* __restrict__ pkBRe, const unsigned* __restrict__ pkBIm, int bIsW)
{
  __shared__ unsigned sAre[128*19];   // [m][k2] packed (K=2k2, 2k2+1), stride 19
  __shared__ unsigned sAim[128*19];
  __shared__ unsigned sBre[16*132];   // [k2][n] packed, stride 132
  __shared__ unsigned sBim[16*132];

  const float2* A = Ab + (long)blockIdx.x * aStride;
  const float2* B = Bb + (long)blockIdx.x * bStride;
  float2*       C = Cb + (long)blockIdx.x * cStride;

  const int tid  = threadIdx.x;
  const int lane = tid & 31;
  const int wave = tid >> 5;
  const int m0   = (wave >> 1) * 16;       // row strip
  const int nCol = (wave & 1) * 4;         // first of 4 column tiles

  v8f accR[4], accI[4];
#pragma unroll
  for (int j = 0; j < 4; ++j) { v8f z = {}; accR[j] = z; accI[j] = z; }

  union U8 { v16h v; unsigned u[8]; };

  for (int kp = 0; kp < 4; ++kp) {
    const int kBase = kp * 32;

#if TDM_ARITY > 0
    if (wave == 0) {   // TDM ignores EXEC; issue once per block from wave 0
      if (aIsW) {      // 128x16dw tile of [128][64] plane, pad 3dw -> stride 19
        tdmLoad2D((unsigned)(size_t)(void*)sAre, pkARe + kp * 16, 64, 128, 16, 128, 64, 3, 2);
        tdmLoad2D((unsigned)(size_t)(void*)sAim, pkAIm + kp * 16, 64, 128, 16, 128, 64, 3, 2);
      }
      if (bIsW) {      // 16x128dw tile of [64][128] plane, pad 4dw -> stride 132
        tdmLoad2D((unsigned)(size_t)(void*)sBre, pkBRe + kp * 16 * 128, 128, 64, 128, 16, 128, 6, 3);
        tdmLoad2D((unsigned)(size_t)(void*)sBim, pkBIm + kp * 16 * 128, 128, 64, 128, 16, 128, 6, 3);
      }
    }
#endif
    if (kp < 3) { // prefetch next K panel of the streamed (non-W) operand
      if (!aIsW) __builtin_prefetch(&A[(tid >> 2) * 128 + kBase + 32], 0, 0);
      if (!bIsW) __builtin_prefetch(&B[(kBase + 32 + (tid >> 7)) * 128 + (tid & 127)], 0, 0);
    }

    if (!aIsW) {        // stage A from f32 data with f16 pair-packing
#pragma unroll
      for (int i = 0; i < 4; ++i) {
        int idx = tid + i * 512;            // 0..2047
        int mm = idx >> 4, ka = idx & 15;
        float2 a0 = A[mm * 128 + kBase + 2 * ka];
        float2 a1 = A[mm * 128 + kBase + 2 * ka + 1];
        sAre[mm * 19 + ka] = packh(a0.x, a1.x);
        sAim[mm * 19 + ka] = packh(a0.y, a1.y);
      }
    }
#if TDM_ARITY == 0
    else {              // packed-plane copy fallback (no TDM)
#pragma unroll
      for (int i = 0; i < 4; ++i) {
        int idx = tid + i * 512;
        int mm = idx >> 4, ka = idx & 15;
        sAre[mm * 19 + ka] = pkARe[mm * 64 + kp * 16 + ka];
        sAim[mm * 19 + ka] = pkAIm[mm * 64 + kp * 16 + ka];
      }
    }
#endif
    if (!bIsW) {        // stage B from f32 data
#pragma unroll
      for (int i = 0; i < 4; ++i) {
        int idx = tid + i * 512;
        int kb = idx >> 7, nn = idx & 127;
        float2 b0 = B[(kBase + 2 * kb) * 128 + nn];
        float2 b1 = B[(kBase + 2 * kb + 1) * 128 + nn];
        sBre[kb * 132 + nn] = packh(b0.x, b1.x);
        sBim[kb * 132 + nn] = packh(b0.y, b1.y);
      }
    }
#if TDM_ARITY == 0
    else {
#pragma unroll
      for (int i = 0; i < 4; ++i) {
        int idx = tid + i * 512;
        int kb = idx >> 7, nn = idx & 127;
        sBre[kb * 132 + nn] = pkBRe[(kp * 16 + kb) * 128 + nn];
        sBim[kb * 132 + nn] = pkBIm[(kp * 16 + kb) * 128 + nn];
      }
    }
#endif
#if TDM_ARITY > 0
    if (wave == 0 && (aIsW || bIsW)) __builtin_amdgcn_s_wait_tensorcnt(0);
#endif
    __syncthreads();

    // A fragment (16-bit A 16x32, contiguous K order, same as B):
    //   lane<16 : M=m0+lane,   K-pairs k2=0..7 ; lane>=16: k2=8..15
    const int Ma  = m0 + (lane & 15);
    const int kA2 = (lane >> 4) * 8;
    U8 far_, fai, fan;
#pragma unroll
    for (int v = 0; v < 8; ++v) {
      int k2 = kA2 + v;
      far_.u[v] = sAre[Ma * 19 + k2];
      fai.u[v]  = sAim[Ma * 19 + k2];
    }
#pragma unroll
    for (int v = 0; v < 8; ++v) fan.u[v] = fai.u[v] ^ 0x80008000u;  // -Ai

    // B fragment (row K striped across lanes):
    //   lane<16 : N=n0+lane, K-pairs k2=0..7 ; lane>=16: k2=8..15
    const int Nn  = lane & 15;
    const int kB2 = (lane >> 4) * 8;
#pragma unroll
    for (int j = 0; j < 4; ++j) {
      const int nb = (nCol + j) * 16 + Nn;
      U8 fbr, fbi;
#pragma unroll
      for (int v = 0; v < 8; ++v) {
        fbr.u[v] = sBre[(kB2 + v) * 132 + nb];
        fbi.u[v] = sBim[(kB2 + v) * 132 + nb];
      }
      // Cr += Ar*Br - Ai*Bi ; Ci += Ar*Bi + Ai*Br   (f32 accumulate)
      accR[j] = __builtin_amdgcn_wmma_f32_16x16x32_f16(false, far_.v, false, fbr.v, (short)0, accR[j], false, false);
      accR[j] = __builtin_amdgcn_wmma_f32_16x16x32_f16(false, fan.v,  false, fbi.v, (short)0, accR[j], false, false);
      accI[j] = __builtin_amdgcn_wmma_f32_16x16x32_f16(false, far_.v, false, fbi.v, (short)0, accI[j], false, false);
      accI[j] = __builtin_amdgcn_wmma_f32_16x16x32_f16(false, fai.v,  false, fbr.v, (short)0, accI[j], false, false);
    }
    __syncthreads();
  }

  // C/D f32 16x16 layout: VGPR r -> M = r + 8*(lane>=16), N = lane&15
  const int half = lane >> 4, nn = lane & 15;
#pragma unroll
  for (int j = 0; j < 4; ++j)
#pragma unroll
    for (int r = 0; r < 8; ++r) {
      int row = m0 + r + 8 * half;
      C[row * 128 + (nCol + j) * 16 + nn] = f2(accR[j][r] * scale, accI[j][r] * scale);
    }
}

// ---------------------------------------------------------------------------
// DFT matrix init: Wf = exp(-2pi i mk/128), Wi = conj (ortho 1/128 applied on
// second GEMM pass).
// ---------------------------------------------------------------------------
__global__ void initW(float2* Wf, float2* Wi)
{
  int idx = blockIdx.x * 256 + threadIdx.x;
  if (idx >= UF) return;
  int m = idx >> 7, k = idx & 127;
  int t = (m * k) & 127;
  float ang = -6.283185307179586f * (float)t / 128.0f;
  float s, c; sincosf(ang, &s, &c);
  Wf[idx] = f2(c, s);
  Wi[idx] = f2(c, -s);
}

// Pre-pack W into f16 K-pair planes matching the GEMM LDS panel layouts:
//   A-side: [m][k2] (64 words/row) ;  B-side: [k2][n] (128 words/row)
__global__ void initWPk(const float2* __restrict__ Wf, const float2* __restrict__ Wi,
                        unsigned* fARe, unsigned* fAIm, unsigned* fBRe, unsigned* fBIm,
                        unsigned* iARe, unsigned* iAIm, unsigned* iBRe, unsigned* iBIm)
{
  int idx = blockIdx.x * 256 + threadIdx.x;
  if (idx >= 128 * 64) return;
  int m = idx >> 6, k2 = idx & 63;       // A view
  float2 a0 = Wf[m * 128 + 2 * k2], a1 = Wf[m * 128 + 2 * k2 + 1];
  fARe[idx] = packh(a0.x, a1.x); fAIm[idx] = packh(a0.y, a1.y);
  float2 c0 = Wi[m * 128 + 2 * k2], c1 = Wi[m * 128 + 2 * k2 + 1];
  iARe[idx] = packh(c0.x, c1.x); iAIm[idx] = packh(c0.y, c1.y);
  int kb = idx >> 7, nn = idx & 127;     // B view (same flat range)
  float2 b0 = Wf[(2 * kb) * 128 + nn], b1 = Wf[(2 * kb + 1) * 128 + nn];
  fBRe[idx] = packh(b0.x, b1.x); fBIm[idx] = packh(b0.y, b1.y);
  float2 d0 = Wi[(2 * kb) * 128 + nn], d1 = Wi[(2 * kb + 1) * 128 + nn];
  iBRe[idx] = packh(d0.x, d1.x); iBIm[idx] = packh(d0.y, d1.y);
}

// sc layout: 0=sigma 1=tau 2=theta 3=rs 4=rs_new 5=alpha 6=beta 7=pAp
__global__ void initScalars(const float* tau_p, const float* sigma_p,
                            const float* theta_p, float* sc)
{
  if (threadIdx.x == 0 && blockIdx.x == 0) {
    float L = sqrtf(13.0f);
    sc[0] = sigmoidf_(sigma_p[0]) / L;
    sc[1] = sigmoidf_(tau_p[0]) / L;
    sc[2] = sigmoidf_(theta_p[0]);
  }
}

__global__ void zeroF(float* p, long n)
{ long i = (long)blockIdx.x * 256 + threadIdx.x; if (i < n) p[i] = 0.f; }

// ---------------------------------------------------------------------------
// Elementwise pipeline kernels
// ---------------------------------------------------------------------------
__global__ void meanT(const float2* __restrict__ y, float2* __restrict__ out)
{
  int idx = blockIdx.x * 256 + threadIdx.x;
  if (idx >= NC * UF) return;
  int c = idx / UF, uf = idx % UF;
  float sr = 0.f, si = 0.f;
  for (int t = 0; t < NT; ++t) { float2 v = y[(c * NT + t) * UF + uf]; sr += v.x; si += v.y; }
  out[idx] = f2(sr / 12.f, si / 12.f);
}

__global__ void packXr(const float2* __restrict__ img, float* __restrict__ xr)
{
  int idx = blockIdx.x * 256 + threadIdx.x;
  if (idx >= NC * UF) return;
  int c = idx / UF, uf = idx % UF;
  float2 v = img[idx];
  xr[(2 * c) * UF + uf] = v.x;
  xr[(2 * c + 1) * UF + uf] = v.y;
}

__global__ void conv2dk(const float* __restrict__ in, const float* __restrict__ w,
                        const float* __restrict__ b, float* __restrict__ out,
                        int CI, int CO, int doRelu)
{
  int idx = blockIdx.x * 256 + threadIdx.x;
  if (idx >= CO * UF) return;
  int o = idx / UF, pos = idx % UF, yy = pos >> 7, xx = pos & 127;
  float acc = b[o];
  for (int ci = 0; ci < CI; ++ci)
    for (int dy = -1; dy <= 1; ++dy) {
      int y2 = yy + dy; if (y2 < 0 || y2 > 127) continue;
      for (int dx = -1; dx <= 1; ++dx) {
        int x2 = xx + dx; if (x2 < 0 || x2 > 127) continue;
        acc += w[((o * CI + ci) * 3 + (dy + 1)) * 3 + (dx + 1)] * in[ci * UF + y2 * 128 + x2];
      }
    }
  out[idx] = doRelu ? fmaxf(acc, 0.f) : acc;
}

__global__ void buildCsm(const float* __restrict__ o, float2* __restrict__ csm)
{
  int idx = blockIdx.x * 256 + threadIdx.x;
  if (idx >= NC * UF) return;
  int c = idx / UF, uf = idx % UF;
  csm[idx] = f2(o[(2 * c) * UF + uf], o[(2 * c + 1) * UF + uf]);
}

__global__ void rssNorm(float2* csm)
{
  int uf = blockIdx.x * 256 + threadIdx.x;
  if (uf >= UF) return;
  float s = 0.f;
  for (int c = 0; c < NC; ++c) { float2 v = csm[c * UF + uf]; s += v.x * v.x + v.y * v.y; }
  float r = 1.f / (sqrtf(s) + 1e-9f);
  for (int c = 0; c < NC; ++c) { float2 v = csm[c * UF + uf]; csm[c * UF + uf] = f2(v.x * r, v.y * r); }
}

__global__ void expandCsm(const float2* __restrict__ csm, const float2* __restrict__ x,
                          float2* __restrict__ out)
{
  long idx = (long)blockIdx.x * 256 + threadIdx.x;
  if (idx >= NY) return;
  int c = (int)(idx / NX); long rem = idx % NX;
  int t = (int)(rem / UF); int uf = (int)(rem % UF);
  out[idx] = cmul(csm[c * UF + uf], x[t * UF + uf]);
}

__global__ void maskApply(float2* __restrict__ dst, const float2* __restrict__ src,
                          const int* __restrict__ mask)
{
  long idx = (long)blockIdx.x * 256 + threadIdx.x;
  if (idx >= NY) return;
  long rem = idx % NX;
  float m = (float)mask[rem];   // mask[t,u,f]
  float2 v = src[idx];
  dst[idx] = f2(v.x * m, v.y * m);
}

__global__ void coilCombine(const float2* __restrict__ csm, const float2* __restrict__ img,
                            float2* __restrict__ out)
{
  int idx = blockIdx.x * 256 + threadIdx.x;
  if (idx >= NX) return;
  int t = idx / UF, uf = idx % UF;
  float2 acc = f2(0.f, 0.f);
  for (int c = 0; c < NC; ++c)
    acc = f2add(acc, cmulConjA(csm[c * UF + uf], img[(c * NT + t) * UF + uf]));
  out[idx] = acc;
}

__global__ void pUpdateK(float2* __restrict__ p, const float2* __restrict__ k,
                         const float2* __restrict__ y, const int* __restrict__ mask,
                         const float* __restrict__ sc)
{
  long idx = (long)blockIdx.x * 256 + threadIdx.x;
  if (idx >= NY) return;
  float m = (float)mask[idx % NX];
  float sig = sc[0], inv = 1.f / (1.f + sc[0]);
  float2 kv = k[idx], yv = y[idx], pv = p[idx];
  p[idx] = f2((pv.x + sig * (kv.x * m - yv.x)) * inv,
              (pv.y + sig * (kv.y * m - yv.y)) * inv);
}

__global__ void qUpdateK(float2* __restrict__ q, const float2* __restrict__ xb,
                         const float* __restrict__ lam, const float* __restrict__ sc)
{
  int idx = blockIdx.x * 256 + threadIdx.x;
  if (idx >= NX) return;
  int t = idx / UF, rem = idx % UF, u = rem >> 7, f = rem & 127;
  float2 xc = xb[idx];
  float2 g[3];
  g[0] = f2sub(xb[t * UF + ((u + 1) & 127) * 128 + f], xc);
  g[1] = f2sub(xb[t * UF + u * 128 + ((f + 1) & 127)], xc);
  g[2] = f2sub(xb[((t + 1) % NT) * UF + u * 128 + f], xc);
  float sig = sc[0];
  for (int d = 0; d < 3; ++d) {
    float2 v = f2add(q[d * NX + idx], f2scale(g[d], sig));
    float a = sqrtf(v.x * v.x + v.y * v.y);
    float l = lam[d * NX + idx];
    float s = l / fmaxf(a, l);
    q[d * NX + idx] = f2scale(v, s);
  }
}

__global__ void ghxK(const float2* __restrict__ xk, const float2* __restrict__ AHp,
                     const float2* __restrict__ q, const float* __restrict__ sc,
                     float2* __restrict__ xn, float2* __restrict__ xb)
{
  int idx = blockIdx.x * 256 + threadIdx.x;
  if (idx >= NX) return;
  int t = idx / UF, rem = idx % UF, u = rem >> 7, f = rem & 127;
  float2 gh = f2(0.f, 0.f);
  gh = f2add(gh, f2sub(q[0 * NX + t * UF + ((u + 127) & 127) * 128 + f], q[0 * NX + idx]));
  gh = f2add(gh, f2sub(q[1 * NX + t * UF + u * 128 + ((f + 127) & 127)], q[1 * NX + idx]));
  gh = f2add(gh, f2sub(q[2 * NX + ((t + NT - 1) % NT) * UF + u * 128 + f], q[2 * NX + idx]));
  float tau = sc[1], th = sc[2];
  float2 xkv = xk[idx];
  float2 x1 = f2sub(xkv, f2scale(f2add(AHp[idx], gh), tau));
  xn[idx] = x1;
  xb[idx] = f2add(x1, f2scale(f2sub(x1, xkv), th));
}

// ------------------------- CG helpers --------------------------------------
__global__ void subK(float* __restrict__ r, const float* __restrict__ b,
                     const float* __restrict__ Ax, long n)
{ long i = (long)blockIdx.x * 256 + threadIdx.x; if (i < n) r[i] = b[i] - Ax[i]; }

__global__ void dotK(const float* __restrict__ a, const float* __restrict__ b,
                     long n, float* __restrict__ out)
{
  __shared__ float red[256];
  float s = 0.f;
  for (long i = (long)blockIdx.x * 256 + threadIdx.x; i < n; i += (long)gridDim.x * 256)
    s += a[i] * b[i];
  red[threadIdx.x] = s; __syncthreads();
  for (int st = 128; st > 0; st >>= 1) {
    if (threadIdx.x < st) red[threadIdx.x] += red[threadIdx.x + st];
    __syncthreads();
  }
  if (threadIdx.x == 0) atomicAdd(out, red[0]);
}

__global__ void cgAlpha(float* sc){ if(!threadIdx.x&&!blockIdx.x) sc[5] = sc[3] / (sc[7] + 1e-12f); }
__global__ void cgBeta (float* sc){ if(!threadIdx.x&&!blockIdx.x){ sc[6] = sc[4] / (sc[3] + 1e-12f); sc[3] = sc[4]; } }

__global__ void cgXR(float* __restrict__ x, float* __restrict__ r,
                     const float* __restrict__ p, const float* __restrict__ Ap,
                     const float* __restrict__ sc, long n)
{
  long i = (long)blockIdx.x * 256 + threadIdx.x; if (i >= n) return;
  float a = sc[5];
  x[i] += a * p[i];
  r[i] -= a * Ap[i];
}

__global__ void cgP(float* __restrict__ p, const float* __restrict__ r,
                    const float* __restrict__ sc, long n)
{
  long i = (long)blockIdx.x * 256 + threadIdx.x; if (i >= n) return;
  p[i] = r[i] + sc[6] * p[i];
}

// ------------------------- lambda_net (conv3d) ------------------------------
__device__ __forceinline__ int reflect128(int i){ if (i < 0) return -i; if (i > 127) return 254 - i; return i; }

#define PU 136
#define PF 136
#define PT 28

__global__ void packP(const float2* __restrict__ x0, float* __restrict__ P)
{
  long idx = (long)blockIdx.x * 256 + threadIdx.x;
  const long tot = 2L * PU * PF * PT;
  if (idx >= tot) return;
  int tt = (int)(idx % PT); long r = idx / PT;
  int ff = (int)(r % PF); r /= PF;
  int uu = (int)(r % PU); int ch = (int)(r / PU);
  int u = reflect128(uu - 4), f = reflect128(ff - 4);
  int t = ((tt - 8) % NT + NT) % NT;
  float2 v = x0[t * UF + u * 128 + f];
  P[idx] = ch ? v.y : v.x;
}

__global__ void conv3d1(const float* __restrict__ P, const float* __restrict__ w,
                        const float* __restrict__ b, float* __restrict__ H)
{
  long idx = (long)blockIdx.x * 256 + threadIdx.x;
  const long tot = 8L * PU * PF * PT;
  if (idx >= tot) return;
  int tt = (int)(idx % PT); long r = idx / PT;
  int ff = (int)(r % PF); r /= PF;
  int uu = (int)(r % PU); int o = (int)(r / PU);
  float acc = b[o];
  for (int ci = 0; ci < 2; ++ci)
    for (int dz = 0; dz < 3; ++dz) {
      int u2 = uu + dz - 1; if (u2 < 0 || u2 >= PU) continue;
      for (int dy = 0; dy < 3; ++dy) {
        int f2i = ff + dy - 1; if (f2i < 0 || f2i >= PF) continue;
        for (int dx = 0; dx < 3; ++dx) {
          int t2 = tt + dx - 1; if (t2 < 0 || t2 >= PT) continue;
          acc += w[(((o * 2 + ci) * 3 + dz) * 3 + dy) * 3 + dx]
               * P[((long)ci * PU + u2) * PF * PT + (long)f2i * PT + t2];
        }
      }
    }
  H[idx] = fmaxf(acc, 0.f);
}

__global__ void conv3d2lam(const float* __restrict__ H, const float* __restrict__ w,
                           const float* __restrict__ b, float* __restrict__ lam)
{
  int idx = blockIdx.x * 256 + threadIdx.x;
  if (idx >= NX) return;
  int t = idx / UF, rem = idx % UF, u = rem >> 7, f = rem & 127;
  int uu = u + 4, ff = f + 4, tt = t + 8;
  float o0 = b[0], o1 = b[1];
  for (int ci = 0; ci < 8; ++ci)
    for (int dz = 0; dz < 3; ++dz)
      for (int dy = 0; dy < 3; ++dy)
        for (int dx = 0; dx < 3; ++dx) {
          float v = H[((long)ci * PU + (uu + dz - 1)) * PF * PT
                      + (long)(ff + dy - 1) * PT + (tt + dx - 1)];
          o0 += w[(((0 * 8 + ci) * 3 + dz) * 3 + dy) * 3 + dx] * v;
          o1 += w[(((1 * 8 + ci) * 3 + dz) * 3 + dy) * 3 + dx] * v;
        }
  float l0 = 1e-6f * softplusf_(o0);
  float l2 = 1e-6f * softplusf_(o1);
  lam[0 * NX + idx] = l0;
  lam[1 * NX + idx] = l0;
  lam[2 * NX + idx] = l2;
}

// ------------------------- outputs -----------------------------------------
__global__ void absK(const float2* __restrict__ x, float* __restrict__ out)
{
  int i = blockIdx.x * 256 + threadIdx.x;
  if (i >= NX) return;
  float2 v = x[i];
  out[i] = sqrtf(v.x * v.x + v.y * v.y);
}

__global__ void xrssK(const float2* __restrict__ img, float* __restrict__ out)
{
  int idx = blockIdx.x * 256 + threadIdx.x;
  if (idx >= NX) return;
  int t = idx / UF, uf = idx % UF;
  float s = 0.f;
  for (int c = 0; c < NC; ++c) {
    float2 v = img[(c * NT + t) * UF + uf];
    s += v.x * v.x + v.y * v.y;
  }
  out[idx] = sqrtf(s);
}

// ---------------------------------------------------------------------------
#define GL(n) dim3((unsigned)(((long)(n) + 255) / 256)), dim3(256), 0, stream

extern "C" void kernel_launch(void* const* d_in, const int* in_sizes, int n_in,
                              void* d_out, int out_size, void* d_ws, size_t ws_size,
                              hipStream_t stream)
{
  (void)in_sizes; (void)n_in; (void)out_size; (void)ws_size;
  const float2* y      = (const float2*)d_in[0];
  const int*    mask   = (const int*)d_in[1];
  const float*  lam_w1 = (const float*)d_in[2];
  const float*  lam_b1 = (const float*)d_in[3];
  const float*  lam_w2 = (const float*)d_in[4];
  const float*  lam_b2 = (const float*)d_in[5];
  const float*  csm_w1 = (const float*)d_in[6];
  const float*  csm_b1 = (const float*)d_in[7];
  const float*  csm_w2 = (const float*)d_in[8];
  const float*  csm_b2 = (const float*)d_in[9];
  const float*  tau_p  = (const float*)d_in[10];
  const float*  sig_p  = (const float*)d_in[11];
  const float*  the_p  = (const float*)d_in[12];

  // bump-allocate workspace
  char* base = (char*)d_ws; size_t off = 0;
  auto alloc = [&](size_t bytes) -> void* {
    void* p = base + off; off = (off + bytes + 255) & ~(size_t)255; return p;
  };
  float2* Wf   = (float2*)alloc((size_t)UF * 8);
  float2* Wi   = (float2*)alloc((size_t)UF * 8);
  unsigned* pk[8];                              // fARe,fAIm,fBRe,fBIm,iARe,iAIm,iBRe,iBIm
  for (int i = 0; i < 8; ++i) pk[i] = (unsigned*)alloc((size_t)128 * 64 * 4);
  float*  sc   = (float*)alloc(64);
  float2* K1   = (float2*)alloc((size_t)NY * 8);
  float2* K2   = (float2*)alloc((size_t)NY * 8);
  float2* pbuf = (float2*)alloc((size_t)NY * 8);
  float2* qbuf = (float2*)alloc((size_t)3 * NX * 8);
  float*  lam  = (float*)alloc((size_t)3 * NX * 4);
  float2* csmB = (float2*)alloc((size_t)NC * UF * 8);
  float2* C1   = (float2*)alloc((size_t)NC * UF * 8);
  float2* C2   = (float2*)alloc((size_t)NC * UF * 8);
  float*  xr   = (float*)alloc((size_t)20 * UF * 4);
  float*  hcn  = (float*)alloc((size_t)16 * UF * 4);
  float*  ocn  = (float*)alloc((size_t)20 * UF * 4);
  float*  Ppad = (float*)alloc(2L * PU * PF * PT * 4);
  float*  Hpad = (float*)alloc(8L * PU * PF * PT * 4);
  float2* AHy  = (float2*)alloc((size_t)NX * 8);
  float2* x0   = (float2*)alloc((size_t)NX * 8);
  float2* xA   = (float2*)alloc((size_t)NX * 8);
  float2* xB   = (float2*)alloc((size_t)NX * 8);
  float2* xb   = (float2*)alloc((size_t)NX * 8);
  float2* tmpx = (float2*)alloc((size_t)NX * 8);
  float2* rb   = (float2*)alloc((size_t)NX * 8);
  float2* pv   = (float2*)alloc((size_t)NX * 8);
  float2* Apb  = (float2*)alloc((size_t)NX * 8);

  const float inv128 = 1.0f / 128.0f;

  // out = (W*src)*W : pass1 DFT along rows (W is A), pass2 along cols (W is B)
  auto runFFT = [&](const float2* src, float2* tmp, float2* dst, int batch,
                    int inv, float scale) {
    const float2* W = inv ? Wi : Wf;
    unsigned *pARe = pk[inv * 4 + 0], *pAIm = pk[inv * 4 + 1];
    unsigned *pBRe = pk[inv * 4 + 2], *pBIm = pk[inv * 4 + 3];
    cgemm128<<<dim3((unsigned)batch), dim3(512), 0, stream>>>(
        W, 0, src, UF, tmp, UF, 1.0f, pARe, pAIm, 1, pBRe, pBIm, 0);
    cgemm128<<<dim3((unsigned)batch), dim3(512), 0, stream>>>(
        tmp, UF, W, 0, dst, UF, scale, pARe, pAIm, 0, pBRe, pBIm, 1);
  };

  auto applyAHA = [&](const float2* xin, float2* xout) {
    expandCsm<<<GL(NY)>>>(csmB, xin, K1);
    runFFT(K1, K2, K1, NSLICE, 0, inv128);
    maskApply<<<GL(NY)>>>(K1, K1, mask);      // m*m == m (binary mask)
    runFFT(K1, K2, K1, NSLICE, 1, inv128);
    coilCombine<<<GL(NX)>>>(csmB, K1, xout);
  };

  // ---- setup ----
  initW<<<GL(UF)>>>(Wf, Wi);
  initWPk<<<GL(128 * 64)>>>(Wf, Wi, pk[0], pk[1], pk[2], pk[3], pk[4], pk[5], pk[6], pk[7]);
  initScalars<<<1, 1, 0, stream>>>(tau_p, sig_p, the_p, sc);

  // ---- csm_net ----
  meanT<<<GL(NC * UF)>>>(y, C1);
  runFFT(C1, C2, C1, NC, 1, inv128);
  packXr<<<GL(NC * UF)>>>(C1, xr);
  conv2dk<<<GL(16 * UF)>>>(xr, csm_w1, csm_b1, hcn, 20, 16, 1);
  conv2dk<<<GL(20 * UF)>>>(hcn, csm_w2, csm_b2, ocn, 16, 20, 0);
  buildCsm<<<GL(NC * UF)>>>(ocn, csmB);
  rssNorm<<<GL(UF)>>>(csmB);

  // ---- AHy and CG warm start ----
  maskApply<<<GL(NY)>>>(K1, y, mask);
  runFFT(K1, K2, K1, NSLICE, 1, inv128);
  coilCombine<<<GL(NX)>>>(csmB, K1, AHy);

  hipMemcpyAsync(x0, AHy, (size_t)NX * 8, hipMemcpyDeviceToDevice, stream);
  applyAHA(x0, Apb);
  subK<<<GL(2L * NX)>>>((float*)rb, (const float*)AHy, (const float*)Apb, 2L * NX);
  hipMemcpyAsync(pv, rb, (size_t)NX * 8, hipMemcpyDeviceToDevice, stream);
  zeroF<<<1, 256, 0, stream>>>(sc + 3, 1);
  dotK<<<dim3(256), dim3(256), 0, stream>>>((const float*)rb, (const float*)rb, 2L * NX, sc + 3);
  for (int it = 0; it < CGIT; ++it) {
    applyAHA(pv, Apb);
    zeroF<<<1, 256, 0, stream>>>(sc + 7, 1);
    dotK<<<dim3(256), dim3(256), 0, stream>>>((const float*)pv, (const float*)Apb, 2L * NX, sc + 7);
    cgAlpha<<<1, 1, 0, stream>>>(sc);
    cgXR<<<GL(2L * NX)>>>((float*)x0, (float*)rb, (const float*)pv, (const float*)Apb, sc, 2L * NX);
    zeroF<<<1, 256, 0, stream>>>(sc + 4, 1);
    dotK<<<dim3(256), dim3(256), 0, stream>>>((const float*)rb, (const float*)rb, 2L * NX, sc + 4);
    cgBeta<<<1, 1, 0, stream>>>(sc);
    cgP<<<GL(2L * NX)>>>((float*)pv, (const float*)rb, sc, 2L * NX);
  }

  // ---- lambda_net ----
  packP<<<GL(2L * PU * PF * PT)>>>(x0, Ppad);
  conv3d1<<<GL(8L * PU * PF * PT)>>>(Ppad, lam_w1, lam_b1, Hpad);
  conv3d2lam<<<GL(NX)>>>(Hpad, lam_w2, lam_b2, lam);

  // ---- PDHG ----
  zeroF<<<GL(2L * NY)>>>((float*)pbuf, 2L * NY);
  zeroF<<<GL(6L * NX)>>>((float*)qbuf, 6L * NX);
  hipMemcpyAsync(xA, x0, (size_t)NX * 8, hipMemcpyDeviceToDevice, stream);
  hipMemcpyAsync(xb, x0, (size_t)NX * 8, hipMemcpyDeviceToDevice, stream);
  float2* xk = xA;
  float2* xn = xB;
  for (int it = 0; it < TPDHG; ++it) {
    // p = (p + sigma*(A(xbar) - y)) / (1+sigma)     [mask fused in pUpdateK]
    expandCsm<<<GL(NY)>>>(csmB, xb, K1);
    runFFT(K1, K2, K1, NSLICE, 0, inv128);
    pUpdateK<<<GL(NY)>>>(pbuf, K1, y, mask, sc);
    // q = clip(q + sigma*G4D(xbar), lam)
    qUpdateK<<<GL(NX)>>>(qbuf, xb, lam, sc);
    // AH(p)
    maskApply<<<GL(NY)>>>(K1, pbuf, mask);
    runFFT(K1, K2, K1, NSLICE, 1, inv128);
    coilCombine<<<GL(NX)>>>(csmB, K1, tmpx);
    // x1 = xk - tau*AH(p) - tau*GH4D(q) ; xbar = x1 + theta*(x1-xk)
    ghxK<<<GL(NX)>>>(xk, tmpx, qbuf, sc, xn, xb);
    float2* t2 = xk; xk = xn; xn = t2;
  }

  // ---- outputs ----
  float* out = (float*)d_out;
  float*  out_px   = out;                              // NX floats
  float2* out_pk   = (float2*)(out + NX);              // NY complex
  float2* out_csm  = (float2*)(out + NX + 2L * NY);    // NC*UF complex
  float*  out_xrss = out + NX + 2L * NY + 2L * NC * UF;// NX floats

  absK<<<GL(NX)>>>(xk, out_px);
  expandCsm<<<GL(NY)>>>(csmB, xk, K1);                 // p_k = A(xk), no mask
  runFFT(K1, K2, out_pk, NSLICE, 0, inv128);
  hipMemcpyAsync(out_csm, csmB, (size_t)NC * UF * 8, hipMemcpyDeviceToDevice, stream);
  runFFT(y, K1, K2, NSLICE, 1, inv128);                // ifft2c(y)
  xrssK<<<GL(NX)>>>(K2, out_xrss);
}
